// GatedBlock_34737695490179
// MI455X (gfx1250) — matrix operand order repaired and checked by reading, not compiled, not run on hardware
//
#include <hip/hip_runtime.h>
#include <hip/hip_bf16.h>
#include <math.h>

typedef __attribute__((ext_vector_type(16))) __bf16 v16bf;
typedef __attribute__((ext_vector_type(4)))  __bf16 v4bf;
typedef __attribute__((ext_vector_type(8)))  float  v8f;

#define BM 128
#define BN 128
#define BKQ 32
#define LDT 40   // LDS row stride in bf16 elems (80B: 16B aligned, 16-row conflict-free)

#define EPI_BIAS           0
#define EPI_GELU           1
#define EPI_SCALE          2
#define EPI_MIX            3
#define EPI_ROWSCALE_BIAS  4

__device__ __forceinline__ float gelu_exact(float x) {
    return 0.5f * x * (1.0f + erff(x * 0.70710678118654752f));
}

// ---------------------------------------------------------------------------
// LDS-tiled bf16 WMMA GEMM:  C[m][n] = sum_k A[m][k] * B[n][k]  (+epilogue)
// A,B are bf16 in global memory. KMAJOR=1: B stored K-major (B[k][n], stride ldb).
// A tiles (both paths) and row-major B tiles are staged with
// GLOBAL_LOAD_ASYNC_TO_LDS_B128 (no data VGPRs, ASYNCcnt-tracked). K-major B
// is register-staged and transposed into LDS. 256 thr = 8 waves; wave grid
// 2(M)x4(N); each wave computes 4x2 v_wmma_f32_16x16x32_bf16 tiles.
// ---------------------------------------------------------------------------
template <int KMAJOR>
__global__ __launch_bounds__(256)
void gemm_bf16_wmma(const __bf16* __restrict__ A, long long lda, long long a_hs,
                    const __bf16* __restrict__ B, long long ldb, long long b_hs,
                    void* __restrict__ C, long long ldc, long long c_hs,
                    int M, int N, int K,
                    int epi, float alpha,
                    const float* __restrict__ bias,
                    const __bf16* __restrict__ aux,
                    const float* __restrict__ gates,
                    int c_bf16)
{
    __shared__ __bf16 As[2][BM * LDT];
    __shared__ __bf16 Bs[2][BM * LDT];

    const int z = blockIdx.z;
    const __bf16* Ab = A + (long long)z * a_hs;
    const __bf16* Bb = B + (long long)z * b_hs;

    const int m0  = blockIdx.y * BM;
    const int n0  = blockIdx.x * BN;
    const int tid = threadIdx.x;
    const int lane = tid & 31;
    const int wid  = tid >> 5;            // 0..7
    const int wm   = (wid >> 2) * 64;     // 0 or 64
    const int wn   = (wid & 3)  * 32;     // 0,32,64,96

    const v8f zacc = {0.f,0.f,0.f,0.f,0.f,0.f,0.f,0.f};
    v8f acc[4][2];
#pragma unroll
    for (int i = 0; i < 4; i++)
#pragma unroll
        for (int j = 0; j < 2; j++) acc[i][j] = zacc;

    // tile indexing: 8 halfs (16B) per thread slot
    const int a_kq = tid & 3;             // 0..3  (k offset = kq*8)
    const int a_r0 = tid >> 2;            // 0..63

    // ---- async memory->LDS staging (no data VGPRs) ----
    auto async_A = [&](int kb, int buf) {
#pragma unroll
        for (int p = 0; p < 2; p++) {
            int row = a_r0 + p * 64;
            int gm  = m0 + row; if (gm > M - 1) gm = M - 1;   // clamp; stores guarded
            unsigned lds = (unsigned)(unsigned long long)(&As[buf][row * LDT + a_kq * 8]);
            unsigned long long ga =
                (unsigned long long)(const void*)(Ab + (long long)gm * lda + kb + a_kq * 8);
            asm volatile("global_load_async_to_lds_b128 %0, %1, off"
                         :: "v"(lds), "v"(ga) : "memory");
        }
    };
    auto async_B = [&](int kb, int buf) {
#pragma unroll
        for (int p = 0; p < 2; p++) {
            int row = a_r0 + p * 64;
            int gn  = n0 + row; if (gn > N - 1) gn = N - 1;
            unsigned lds = (unsigned)(unsigned long long)(&Bs[buf][row * LDT + a_kq * 8]);
            unsigned long long ga =
                (unsigned long long)(const void*)(Bb + (long long)gn * ldb + kb + a_kq * 8);
            asm volatile("global_load_async_to_lds_b128 %0, %1, off"
                         :: "v"(lds), "v"(ga) : "memory");
        }
    };

    // ---- register-staged transpose loader for K-major B (small GEMM only) --
    uint2 sbk[4];
    const int bk_cg = tid & 31;           // column group (4 cols)
    const int bk_r0 = tid >> 5;           // 0..7 (k row base)
    auto fetch_kmB = [&](int kb) {
#pragma unroll
        for (int p = 0; p < 4; p++) {
            int k   = bk_r0 + p * 8;
            int gn0 = n0 + bk_cg * 4;
            const __bf16* src = Bb + (long long)(kb + k) * ldb;
            if (gn0 + 3 <= N - 1) {
                sbk[p] = *(const uint2*)(src + gn0);
            } else {
                __bf16 t[4];
                int c0 = gn0     < N ? gn0     : N - 1;
                int c1 = gn0 + 1 < N ? gn0 + 1 : N - 1;
                int c2 = gn0 + 2 < N ? gn0 + 2 : N - 1;
                int c3 = gn0 + 3 < N ? gn0 + 3 : N - 1;
                t[0] = src[c0]; t[1] = src[c1]; t[2] = src[c2]; t[3] = src[c3];
                sbk[p] = *(const uint2*)t;
            }
        }
    };
    auto commit_kmB = [&](int buf) {
#pragma unroll
        for (int p = 0; p < 4; p++) {
            int k = bk_r0 + p * 8;
            __bf16 t[4];
            *(uint2*)t = sbk[p];
            int colb = bk_cg * 4;
            Bs[buf][(colb + 0) * LDT + k] = t[0];
            Bs[buf][(colb + 1) * LDT + k] = t[1];
            Bs[buf][(colb + 2) * LDT + k] = t[2];
            Bs[buf][(colb + 3) * LDT + k] = t[3];
        }
    };

    // ---- compute one K-step from LDS buffer 'buf' ----
    // Fragment layouts per CDNA5 ISA:
    //   A 16x32: lane<16 -> row=lane, K {0..7,16..23}; lane>=16 -> row-16, K {8..15,24..31}
    //   B 32x16: lane<16 -> col=lane, K {0..7,16..23}; lane>=16 -> col-16, K {8..15,24..31}
    const int arow = lane & 15;
    const int ak0  = (lane >> 4) * 8;
    const int bk0  = (lane >> 4) * 8;
    auto compute = [&](int buf) {
        union Frag { v16bf v; uint4 u[2]; };
        Frag afrag[4], bfrag[2];
#pragma unroll
        for (int i = 0; i < 4; i++) {
            const __bf16* p = &As[buf][(wm + i * 16 + arow) * LDT + ak0];
            afrag[i].u[0] = *(const uint4*)(p);
            afrag[i].u[1] = *(const uint4*)(p + 16);
        }
#pragma unroll
        for (int j = 0; j < 2; j++) {
            const __bf16* p = &Bs[buf][(wn + j * 16 + arow) * LDT + bk0];
            bfrag[j].u[0] = *(const uint4*)(p);
            bfrag[j].u[1] = *(const uint4*)(p + 16);
        }
#pragma unroll
        for (int i = 0; i < 4; i++)
#pragma unroll
            for (int j = 0; j < 2; j++)
                acc[i][j] = __builtin_amdgcn_wmma_f32_16x16x32_bf16(
                    false, afrag[i].v, false, bfrag[j].v,
                    (short)0, acc[i][j], false, false);
    };

    int buf = 0;
    if (!KMAJOR) {
        async_A(0, 0);
        async_B(0, 0);
        asm volatile("s_wait_asynccnt 0x0" ::: "memory");
        __syncthreads();
        for (int kb = 0; kb < K; kb += BKQ) {
            const bool hasNext = (kb + BKQ < K);
            if (hasNext) { async_A(kb + BKQ, buf ^ 1); async_B(kb + BKQ, buf ^ 1); }
            compute(buf);
            asm volatile("s_wait_asynccnt 0x0" ::: "memory");
            __syncthreads();
            buf ^= 1;
        }
    } else {
        async_A(0, 0);
        fetch_kmB(0);
        asm volatile("s_wait_asynccnt 0x0" ::: "memory");
        commit_kmB(0);
        __syncthreads();
        for (int kb = 0; kb < K; kb += BKQ) {
            const bool hasNext = (kb + BKQ < K);
            if (hasNext) { async_A(kb + BKQ, buf ^ 1); fetch_kmB(kb + BKQ); }
            compute(buf);
            asm volatile("s_wait_asynccnt 0x0" ::: "memory");
            __syncthreads();                 // reads of buf^1 done; A DMA landed
            if (hasNext) commit_kmB(buf ^ 1);
            __syncthreads();                 // B transpose visible
            buf ^= 1;
        }
    }

    // C layout: lane 0-15 -> N=lane, M=e; lane 16-31 -> N=lane-16, M=e+8
    float*  Cf = (float*)C  + (long long)z * c_hs;
    __bf16* Ch = (__bf16*)C + (long long)z * c_hs;
    const int col   = lane & 15;
    const int rbase = (lane >> 4) * 8;
#pragma unroll
    for (int i = 0; i < 4; i++) {
#pragma unroll
        for (int j = 0; j < 2; j++) {
            const int gn = n0 + wn + j * 16 + col;
#pragma unroll
            for (int e = 0; e < 8; e++) {
                const int gm = m0 + wm + i * 16 + rbase + e;
                if (gm < M && gn < N) {
                    float v = acc[i][j][e];
                    if (epi == EPI_BIAS) {
                        if (bias) v += bias[gn];
                    } else if (epi == EPI_GELU) {
                        v = gelu_exact(v + (bias ? bias[gn] : 0.0f));
                    } else if (epi == EPI_SCALE) {
                        v *= alpha;
                    } else if (epi == EPI_MIX) {
                        const float g0 = gates[2 * gm], g1 = gates[2 * gm + 1];
                        v = g0 * (float)aux[(long long)gm * ldc + gn]
                          + g1 * gelu_exact(v + bias[gn]);
                    } else { // EPI_ROWSCALE_BIAS
                        v += (gates[2 * gm] + gates[2 * gm + 1]) * bias[gn];
                    }
                    const long long idx = (long long)gm * ldc + gn;
                    if (c_bf16) Ch[idx] = (__bf16)v;
                    else        Cf[idx] = v;
                }
            }
        }
    }
}

// ---------------------------------------------------------------------------
// f32 -> bf16 elementwise convert (weights / x), vectorized x4.
// ---------------------------------------------------------------------------
__global__ __launch_bounds__(256)
void cvt_bf16_kernel(const float* __restrict__ src, __bf16* __restrict__ dst, long long n)
{
    const long long i = ((long long)blockIdx.x * 256 + threadIdx.x) * 4;
    if (i + 3 < n) {
        const float4 v = *(const float4*)(src + i);
        v4bf p;
        p.x = (__bf16)v.x; p.y = (__bf16)v.y; p.z = (__bf16)v.z; p.w = (__bf16)v.w;
        *(v4bf*)(dst + i) = p;
    } else {
        for (long long j = i; j < n; j++) dst[j] = (__bf16)src[j];
    }
}

// ---------------------------------------------------------------------------
// LayerNorm over last dim D; writes bf16 y and/or bf16 (y + pos[row%S]).
// ---------------------------------------------------------------------------
__global__ __launch_bounds__(256)
void ln_kernel(const float* __restrict__ x, const float* __restrict__ g,
               const float* __restrict__ b, const float* __restrict__ pos,
               __bf16* __restrict__ out, __bf16* __restrict__ out_pe,
               int D, int S)
{
    __shared__ float red[256];
    const long long row = blockIdx.x;
    const float* xr = x + row * D;
    const int tid = threadIdx.x;

    float s = 0.f;
    for (int c = tid; c < D; c += 256) s += xr[c];
    red[tid] = s; __syncthreads();
    for (int off = 128; off > 0; off >>= 1) { if (tid < off) red[tid] += red[tid + off]; __syncthreads(); }
    const float mu = red[0] / (float)D;
    __syncthreads();

    float v = 0.f;
    for (int c = tid; c < D; c += 256) { float d = xr[c] - mu; v += d * d; }
    red[tid] = v; __syncthreads();
    for (int off = 128; off > 0; off >>= 1) { if (tid < off) red[tid] += red[tid + off]; __syncthreads(); }
    const float rstd = rsqrtf(red[0] / (float)D + 1e-5f);

    const int sr = (int)(row % S);
    for (int c = tid; c < D; c += 256) {
        float y = (xr[c] - mu) * rstd * g[c] + b[c];
        if (out)    out[row * D + c]    = (__bf16)y;
        if (out_pe) out_pe[row * D + c] = (__bf16)(y + pos[(long long)sr * D + c]);
    }
}

// ---------------------------------------------------------------------------
// Row softmax: read f32 scores, write bf16 probabilities (ncols <= 2048).
// ---------------------------------------------------------------------------
__global__ __launch_bounds__(256)
void softmax_kernel(const float* __restrict__ S, __bf16* __restrict__ P, int ncols)
{
    __shared__ float red[256];
    const float* p = S + (long long)blockIdx.x * ncols;
    __bf16*      q = P + (long long)blockIdx.x * ncols;
    const int tid = threadIdx.x;

    float ev[8];
    float m = -3.4e38f;
    for (int c = tid; c < ncols; c += 256) m = fmaxf(m, p[c]);
    red[tid] = m; __syncthreads();
    for (int off = 128; off > 0; off >>= 1) { if (tid < off) red[tid] = fmaxf(red[tid], red[tid + off]); __syncthreads(); }
    m = red[0]; __syncthreads();

    float s = 0.f;
    int nq = 0;
    for (int c = tid; c < ncols; c += 256) { float e = expf(p[c] - m); ev[nq++] = e; s += e; }
    red[tid] = s; __syncthreads();
    for (int off = 128; off > 0; off >>= 1) { if (tid < off) red[tid] += red[tid + off]; __syncthreads(); }
    const float inv = 1.0f / red[0];
    nq = 0;
    for (int c = tid; c < ncols; c += 256) q[c] = (__bf16)(ev[nq++] * inv);
}

// ---------------------------------------------------------------------------
// 2-expert noisy gating. With K=2 experts, top_k keeps both; gates ~= softmax.
// ---------------------------------------------------------------------------
__global__ __launch_bounds__(256)
void gating_kernel(const float* __restrict__ x, const float* __restrict__ wg,
                   const float* __restrict__ wn, const float* __restrict__ noise,
                   float* __restrict__ gates, int D)
{
    __shared__ float red[256][4];
    const long long r = blockIdx.x;
    const float* xr = x + r * D;
    const int tid = threadIdx.x;

    float a0 = 0.f, a1 = 0.f, n0 = 0.f, n1 = 0.f;
    for (int c = tid; c < D; c += 256) {
        const float xv = xr[c];
        a0 += xv * wg[c * 2];  a1 += xv * wg[c * 2 + 1];
        n0 += xv * wn[c * 2];  n1 += xv * wn[c * 2 + 1];
    }
    red[tid][0] = a0; red[tid][1] = a1; red[tid][2] = n0; red[tid][3] = n1;
    __syncthreads();
    for (int off = 128; off > 0; off >>= 1) {
        if (tid < off) { for (int q = 0; q < 4; q++) red[tid][q] += red[tid + off][q]; }
        __syncthreads();
    }
    if (tid == 0) {
        const float c0 = red[0][0], c1 = red[0][1];
        const float s0 = (red[0][2] > 20.f ? red[0][2] : log1pf(expf(red[0][2]))) + 0.01f;
        const float s1 = (red[0][3] > 20.f ? red[0][3] : log1pf(expf(red[0][3]))) + 0.01f;
        const float l0 = c0 + noise[2 * r]     * s0;
        const float l1 = c1 + noise[2 * r + 1] * s1;
        const float mm = fmaxf(l0, l1);
        const float e0 = expf(l0 - mm), e1 = expf(l1 - mm);
        const float p0 = e0 / (e0 + e1), p1 = e1 / (e0 + e1);
        const float inv = 1.0f / (p0 + p1 + 1e-6f);
        gates[2 * r]     = p0 * inv;
        gates[2 * r + 1] = p1 * inv;
    }
}

// ---------------------------------------------------------------------------
extern "C" void kernel_launch(void* const* d_in, const int* in_sizes, int n_in,
                              void* d_out, int out_size, void* d_ws, size_t ws_size,
                              hipStream_t stream)
{
    const float* x    = (const float*)d_in[0];
    const float* nois = (const float*)d_in[1];
    const float* W1   = (const float*)d_in[2];
    const float* b1   = (const float*)d_in[3];
    const float* W2   = (const float*)d_in[4];
    const float* b2   = (const float*)d_in[5];
    const float* query= (const float*)d_in[6];
    const float* pos  = (const float*)d_in[7];
    const float* lnqg = (const float*)d_in[8];
    const float* lnqb = (const float*)d_in[9];
    const float* lnkg = (const float*)d_in[10];
    const float* lnkb = (const float*)d_in[11];
    const float* Wqkv = (const float*)d_in[12];
    const float* bqkv = (const float*)d_in[13];
    const float* Wo   = (const float*)d_in[14];
    const float* bo   = (const float*)d_in[15];
    const float* wg   = (const float*)d_in[16];
    const float* wn   = (const float*)d_in[17];
    float* out = (float*)d_out;

    const int B = 16, S = 576, D = 1024, H = 8, DH = 128, HID = 4096;
    const long long ND = (long long)B * S;   // 9216 rows

    // ---- workspace (bf16 operands; ~220 MB). d_out doubles as scratch. ----
    char* wsb = (char*)d_ws;
    auto alloc = [&](long long elems, int esz) {
        void* p = wsb;
        wsb += ((long long)elems * esz + 255) & ~255LL;
        return p;
    };
    __bf16* Wqkv_h = (__bf16*)alloc(3LL * D * D, 2);
    __bf16* Wo_h   = (__bf16*)alloc((long long)D * D, 2);
    __bf16* W1_h   = (__bf16*)alloc((long long)HID * D, 2);
    __bf16* W2_h   = (__bf16*)alloc((long long)HID * HID, 2);
    __bf16* x_h    = (__bf16*)alloc(ND * D, 2);
    __bf16* kpe_h  = (__bf16*)alloc(ND * D, 2);
    __bf16* qln_h  = (__bf16*)alloc((long long)S * D, 2);
    __bf16* qp_h   = (__bf16*)alloc((long long)S * D, 2);
    __bf16* kp_h   = (__bf16*)alloc(ND * D, 2);
    __bf16* vp_h   = (__bf16*)alloc(ND * D, 2);
    float*  Sb     = (float*)alloc((long long)H * S * S, 4);   // per-batch scores (f32)
    __bf16* probs_h= (__bf16*)alloc((long long)H * S * S, 2);
    __bf16* att_h  = (__bf16*)alloc(ND * D, 2);                // out-proj result
    __bf16* h0_h   = (__bf16*)alloc(ND * HID, 2);              // gelu(x@W1^T), then hmix
    float*  gates  = (float*)alloc(ND * 2, 4);
    // d_out reuse (disjoint lifetimes): kv bf16 then per-head attention out bf16
    __bf16* kv_h    = (__bf16*)out;   // consumed by V projection (before batch loop)
    __bf16* attnO_h = (__bf16*)out;   // written in batch loop, consumed by out-proj

    auto gemm = [&](const __bf16* A, long long lda, long long ahs,
                    const __bf16* Bm, long long ldb, long long bhs,
                    void* C, long long ldc, long long chs,
                    int M, int N, int K, int nz,
                    int epi, float alpha, const float* bias,
                    const __bf16* aux, const float* g, int kmaj, int cbf) {
        dim3 grid((N + BN - 1) / BN, (M + BM - 1) / BM, nz);
        if (kmaj)
            gemm_bf16_wmma<1><<<grid, 256, 0, stream>>>(A, lda, ahs, Bm, ldb, bhs, C, ldc, chs,
                                                        M, N, K, epi, alpha, bias, aux, g, cbf);
        else
            gemm_bf16_wmma<0><<<grid, 256, 0, stream>>>(A, lda, ahs, Bm, ldb, bhs, C, ldc, chs,
                                                        M, N, K, epi, alpha, bias, aux, g, cbf);
    };
    auto cvt = [&](const float* s, __bf16* d, long long n) {
        long long blocks = (n / 4 + 255) / 256 + 1;
        cvt_bf16_kernel<<<dim3((unsigned)blocks), 256, 0, stream>>>(s, d, n);
    };

    // One-time operand conversions (weights + x)
    cvt(Wqkv, Wqkv_h, 3LL * D * D);
    cvt(Wo,   Wo_h,   (long long)D * D);
    cvt(W1,   W1_h,   (long long)HID * D);
    cvt(W2,   W2_h,   (long long)HID * HID);
    cvt(x,    x_h,    ND * D);

    // Gating (independent of attention path)
    gating_kernel<<<dim3((unsigned)ND), 256, 0, stream>>>(x, wg, wn, nois, gates, D);

    // LayerNorms (+pos_embed), bf16 outputs
    ln_kernel<<<dim3((unsigned)ND), 256, 0, stream>>>(x, lnkg, lnkb, pos, kv_h, kpe_h, D, S);
    ln_kernel<<<dim3(S), 256, 0, stream>>>(query, lnqg, lnqb, pos, nullptr, qln_h, D, S);

    // QKV projections (q is batch-invariant -> computed once)
    gemm(qln_h, D, 0, Wqkv_h,                    D, 0, qp_h, D, 0, S,       D, D, 1, EPI_BIAS, 0.f, bqkv,          nullptr, nullptr, 0, 1);
    gemm(kpe_h, D, 0, Wqkv_h + (long long)D * D, D, 0, kp_h, D, 0, (int)ND, D, D, 1, EPI_BIAS, 0.f, bqkv + D,     nullptr, nullptr, 0, 1);
    gemm(kv_h,  D, 0, Wqkv_h + 2LL * D * D,      D, 0, vp_h, D, 0, (int)ND, D, D, 1, EPI_BIAS, 0.f, bqkv + 2 * D, nullptr, nullptr, 0, 1);

    // Attention, per batch (heads batched on grid.z)
    const float scale = 0.08838834764831845f;  // 1/sqrt(128)
    for (int b = 0; b < B; b++) {
        gemm(qp_h, D, DH,
             kp_h + (long long)b * S * D, D, DH,
             Sb, S, (long long)S * S,
             S, S, DH, H, EPI_SCALE, scale, nullptr, nullptr, nullptr, 0, 0);
        softmax_kernel<<<dim3(H * S), 256, 0, stream>>>(Sb, probs_h, S);
        gemm(probs_h, S, (long long)S * S,
             vp_h + (long long)b * S * D, D, DH,            // K-major (rows = seq)
             attnO_h + (long long)b * S * D, D, DH,
             S, DH, S, H, EPI_BIAS, 0.f, nullptr, nullptr, nullptr, 1, 1);
    }

    // Output projection
    gemm(attnO_h, D, 0, Wo_h, D, 0, att_h, D, 0, (int)ND, D, D, 1, EPI_BIAS, 0.f, bo, nullptr, nullptr, 0, 1);

    // Expert path. Per-row gates commute with the second projection:
    //   g0*(h0@W2^T) + g1*(h1@W2^T) = (g0*h0 + g1*h1)@W2^T + (g0+g1)*b2
    gemm(x_h,   D, 0, W1_h, D, 0, h0_h, HID, 0, (int)ND, HID, D, 1, EPI_GELU, 0.f, b1, nullptr, nullptr, 0, 1);
    gemm(att_h, D, 0, W1_h, D, 0, h0_h, HID, 0, (int)ND, HID, D, 1, EPI_MIX,  0.f, b1, h0_h, gates, 0, 1);
    gemm(h0_h, HID, 0, W2_h, HID, 0, out, HID, 0, (int)ND, HID, HID, 1, EPI_ROWSCALE_BIAS, 0.f, b2, nullptr, gates, 0, 0);
}
